// CodeGNN_16578573763454
// MI455X (gfx1250) — matrix operand order repaired and checked by reading, not compiled, not run on hardware
//
#include <hip/hip_runtime.h>
#include <hip/hip_bf16.h>

typedef __attribute__((ext_vector_type(16))) __bf16 v16bf;
typedef __attribute__((ext_vector_type(8)))  float  v8f;

#define HID   64
#define OUTC  128
#define NGR   128

// ---------------------------------------------------------------------------
// helpers
// ---------------------------------------------------------------------------
__device__ inline v16bf zero_frag() {
    v16bf z;
#pragma unroll
    for (int j = 0; j < 16; ++j) z[j] = (__bf16)0.0f;
    return z;
}

// Build one 16x32 bf16 A-fragment (CDNA5 layout:
// lanes 0-15 hold row M=lane, K = lo..lo+7 (v0-3) and lo+16..lo+23 (v4-7),
// where lo = (lane>=16)*8). rowp points at this lane's row, scale applied.
template <int KIN>
__device__ inline v16bf load_A_frag(const float* __restrict__ rowp, int kbase,
                                    int lo, bool valid, float scale) {
    v16bf a;
    if (KIN >= 32 && valid) {
        // full-chunk vector path: 4x float4 loads
        const float4* p0 = (const float4*)(rowp + kbase + lo);
        const float4* p1 = (const float4*)(rowp + kbase + lo + 16);
        float4 f0 = p0[0], f1 = p0[1], f2 = p1[0], f3 = p1[1];
        a[0]  = (__bf16)(f0.x * scale); a[1]  = (__bf16)(f0.y * scale);
        a[2]  = (__bf16)(f0.z * scale); a[3]  = (__bf16)(f0.w * scale);
        a[4]  = (__bf16)(f1.x * scale); a[5]  = (__bf16)(f1.y * scale);
        a[6]  = (__bf16)(f1.z * scale); a[7]  = (__bf16)(f1.w * scale);
        a[8]  = (__bf16)(f2.x * scale); a[9]  = (__bf16)(f2.y * scale);
        a[10] = (__bf16)(f2.z * scale); a[11] = (__bf16)(f2.w * scale);
        a[12] = (__bf16)(f3.x * scale); a[13] = (__bf16)(f3.y * scale);
        a[14] = (__bf16)(f3.z * scale); a[15] = (__bf16)(f3.w * scale);
    } else {
#pragma unroll
        for (int j = 0; j < 16; ++j) {
            int k = kbase + lo + (j < 8 ? j : j + 8);
            float v = (valid && k < KIN) ? rowp[k] * scale : 0.0f;
            a[j] = (__bf16)v;
        }
    }
    return a;
}

// ---------------------------------------------------------------------------
// Pack a [K x Ncols] f32 weight into WMMA bf16 B-fragments.
// B 32x16 layout: col=lane%16, K=ktbase+lo+(j<8 ? j : j+8), lo=(lane>=16)*8.
// out[((kt*nnt+nt)*32 + lane)*16 + j]
// ---------------------------------------------------------------------------
__global__ void pack_weight_kern(const float* __restrict__ W,
                                 __bf16* __restrict__ out, int K, int Ncols) {
    int idx = blockIdx.x * blockDim.x + threadIdx.x;
    int nnt = Ncols >> 4;
    int total = 2 * nnt * 32 * 16;
    if (idx >= total) return;
    int j    = idx & 15;
    int lane = (idx >> 4) & 31;
    int frag = idx >> 9;
    int kt = frag / nnt;
    int nt = frag % nnt;
    int col = (nt << 4) + (lane & 15);
    int k   = (kt << 5) + ((lane >> 4) << 3) + (j < 8 ? j : j + 8);
    float v = (k < K) ? W[k * Ncols + col] : 0.0f;
    out[idx] = (__bf16)v;
}

// ---------------------------------------------------------------------------
// Degree / inverse degree
// ---------------------------------------------------------------------------
__global__ void deg_kern(const int* __restrict__ dst, float* __restrict__ cnt,
                         int E) {
    int e = blockIdx.x * blockDim.x + threadIdx.x;
    if (e >= E) return;
    atomicAdd(cnt + dst[e], 1.0f);
}

__global__ void invdeg_kern(float* __restrict__ cnt, int n) {
    int i = blockIdx.x * blockDim.x + threadIdx.x;
    if (i >= n) return;
    cnt[i] = 1.0f / fmaxf(cnt[i], 1.0f);
}

// ---------------------------------------------------------------------------
// Edge scatter-add (mean numerator). Node arrays (25.6MB) are L2-resident on
// MI455X (192MB L2), so flat f32 atomics resolve in L2 at high rate.
// ---------------------------------------------------------------------------
__global__ void scatter3_kern(const float* __restrict__ x,
                              const int* __restrict__ src,
                              const int* __restrict__ dst,
                              float* __restrict__ agg, int E) {
    int e = blockIdx.x * blockDim.x + threadIdx.x;
    if (e >= E) return;
    int s = src[e], d = dst[e];
    const float* xs = x + (size_t)s * 3;
    float* a = agg + (size_t)d * 3;
    atomicAdd(a + 0, xs[0]);
    atomicAdd(a + 1, xs[1]);
    atomicAdd(a + 2, xs[2]);
}

__global__ void scatter64_kern(const float* __restrict__ h,
                               const int* __restrict__ src,
                               const int* __restrict__ dst,
                               float* __restrict__ agg, int E) {
    long long idx = (long long)blockIdx.x * blockDim.x + threadIdx.x;
    if (idx >= (long long)E * 16) return;
    int e = (int)(idx >> 4);
    int c = ((int)idx & 15) << 2;
    int s = src[e], d = dst[e];
    float4 v = *(const float4*)(h + (size_t)s * HID + c);
    float* a = agg + (size_t)d * HID + c;
    atomicAdd(a + 0, v.x);
    atomicAdd(a + 1, v.y);
    atomicAdd(a + 2, v.z);
    atomicAdd(a + 3, v.w);
}

// ---------------------------------------------------------------------------
// SAGE node update:  hout = relu( (agg*invdeg) @ Wl + bl + hin @ Wr )
// One wave per 16-row tile; bf16 WMMA 16x16x32, f32 accumulate.
// KIN is a template parameter so the K-chunk structure is compile-time:
// no register arrays with runtime indices (avoids v_movrels), straight-line
// WMMA sequence (16 for KIN=64, 8 for KIN=3).
// ---------------------------------------------------------------------------
template <int KIN>
__global__ __launch_bounds__(256) void sage_layer_kern(
    const float* __restrict__ hin, const float* __restrict__ agg,
    const float* __restrict__ invc, const __bf16* __restrict__ Wlp,
    const __bf16* __restrict__ Wrp, const float* __restrict__ bias,
    float* __restrict__ hout, int n) {
    constexpr int NKT = (KIN + 31) / 32;  // 1 for KIN=3, 2 for KIN=64
    int gtid = blockIdx.x * blockDim.x + threadIdx.x;
    int wave = gtid >> 5;
    int lane = threadIdx.x & 31;
    int r0 = wave << 4;
    if (r0 >= n) return;
    int m  = lane & 15;
    int hi = lane >> 4;
    int lo = hi << 3;
    int row = r0 + m;
    bool valid = row < n;
    int safe = valid ? row : 0;
    const float* xrow = hin + (size_t)safe * KIN;
    const float* arow = agg + (size_t)safe * KIN;
    float inv = valid ? invc[safe] : 0.0f;

    v16bf am0 = load_A_frag<KIN>(arow, 0, lo, valid, inv);
    v16bf ax0 = load_A_frag<KIN>(xrow, 0, lo, valid, 1.0f);
    v16bf am1 = zero_frag();
    v16bf ax1 = zero_frag();
    if (NKT > 1) {
        am1 = load_A_frag<KIN>(arow, 32, lo, valid, inv);
        ax1 = load_A_frag<KIN>(xrow, 32, lo, valid, 1.0f);
    }

#pragma unroll
    for (int nt = 0; nt < 4; ++nt) {
        v8f acc = {};
        {
            v16bf bl = *(const v16bf*)(Wlp + (size_t)(nt * 32 + lane) * 16);
            acc = __builtin_amdgcn_wmma_f32_16x16x32_bf16(
                false, am0, false, bl, (short)0, acc, false, false);
            v16bf br = *(const v16bf*)(Wrp + (size_t)(nt * 32 + lane) * 16);
            acc = __builtin_amdgcn_wmma_f32_16x16x32_bf16(
                false, ax0, false, br, (short)0, acc, false, false);
        }
        if (NKT > 1) {
            v16bf bl = *(const v16bf*)(Wlp + (size_t)((4 + nt) * 32 + lane) * 16);
            acc = __builtin_amdgcn_wmma_f32_16x16x32_bf16(
                false, am1, false, bl, (short)0, acc, false, false);
            v16bf br = *(const v16bf*)(Wrp + (size_t)((4 + nt) * 32 + lane) * 16);
            acc = __builtin_amdgcn_wmma_f32_16x16x32_bf16(
                false, ax1, false, br, (short)0, acc, false, false);
        }
        float b = bias[(nt << 4) + m];
#pragma unroll
        for (int j = 0; j < 8; ++j) {
            int orow = r0 + (hi << 3) + j;
            if (orow < n) {
                float v = acc[j] + b;
                hout[(size_t)orow * HID + (nt << 4) + m] = v > 0.0f ? v : 0.0f;
            }
        }
    }
}

// ---------------------------------------------------------------------------
// Global mean pool (numerator + count)
// ---------------------------------------------------------------------------
__global__ void pool_kern(const float* __restrict__ h,
                          const int* __restrict__ batch,
                          float* __restrict__ psum, float* __restrict__ pcnt,
                          int n) {
    long long idx = (long long)blockIdx.x * blockDim.x + threadIdx.x;
    if (idx >= (long long)n * 16) return;
    int node = (int)(idx >> 4);
    int c = ((int)idx & 15) << 2;
    int g = batch[node] & (NGR - 1);
    float4 v = *(const float4*)(h + (size_t)node * HID + c);
    float* p = psum + (size_t)g * HID + c;
    atomicAdd(p + 0, v.x);
    atomicAdd(p + 1, v.y);
    atomicAdd(p + 2, v.z);
    atomicAdd(p + 3, v.w);
    if (((int)idx & 15) == 0) atomicAdd(pcnt + g, 1.0f);
}

// ---------------------------------------------------------------------------
// Final FC + row L2-normalize.  G=128 rows -> exactly 8 waves x 16 rows.
// Each wave owns full rows, so the norm is a pure shfl_xor reduction over the
// 16-lane half-waves that share each row.  No LDS needed.
// ---------------------------------------------------------------------------
__global__ __launch_bounds__(256) void final_fc_kern(
    const float* __restrict__ psum, const float* __restrict__ pcnt,
    const __bf16* __restrict__ Wfcp, const float* __restrict__ bfc,
    float* __restrict__ out) {
    int wave = threadIdx.x >> 5;
    int lane = threadIdx.x & 31;
    int r0 = wave << 4;
    int m = lane & 15;
    int hi = lane >> 4;
    int lo = hi << 3;
    int row = r0 + m;
    float inv = 1.0f / fmaxf(pcnt[row], 1.0f);
    const float* prow = psum + (size_t)row * HID;

    v16bf a0 = load_A_frag<HID>(prow, 0, lo, true, inv);
    v16bf a1 = load_A_frag<HID>(prow, 32, lo, true, inv);

    v8f accs[8];
#pragma unroll
    for (int nt = 0; nt < 8; ++nt) {
        v8f acc = {};
        {
            v16bf b = *(const v16bf*)(Wfcp + (size_t)(nt * 32 + lane) * 16);
            acc = __builtin_amdgcn_wmma_f32_16x16x32_bf16(
                false, a0, false, b, (short)0, acc, false, false);
        }
        {
            v16bf b = *(const v16bf*)(Wfcp + (size_t)((8 + nt) * 32 + lane) * 16);
            acc = __builtin_amdgcn_wmma_f32_16x16x32_bf16(
                false, a1, false, b, (short)0, acc, false, false);
        }
        float bb = bfc[(nt << 4) + m];
#pragma unroll
        for (int j = 0; j < 8; ++j) acc[j] += bb;
        accs[nt] = acc;
    }

#pragma unroll
    for (int j = 0; j < 8; ++j) {
        float ss = 0.0f;
#pragma unroll
        for (int nt = 0; nt < 8; ++nt) ss += accs[nt][j] * accs[nt][j];
        // reduce across the 16 lanes holding this row (wave32, halves disjoint)
        ss += __shfl_xor(ss, 1, 32);
        ss += __shfl_xor(ss, 2, 32);
        ss += __shfl_xor(ss, 4, 32);
        ss += __shfl_xor(ss, 8, 32);
        float norm = fmaxf(sqrtf(ss), 1e-12f);
        int orow = r0 + (hi << 3) + j;
#pragma unroll
        for (int nt = 0; nt < 8; ++nt)
            out[(size_t)orow * OUTC + (nt << 4) + m] = accs[nt][j] / norm;
    }
}

// ---------------------------------------------------------------------------
// Launcher
// ---------------------------------------------------------------------------
extern "C" void kernel_launch(void* const* d_in, const int* in_sizes, int n_in,
                              void* d_out, int out_size, void* d_ws,
                              size_t ws_size, hipStream_t stream) {
    (void)n_in; (void)out_size; (void)ws_size;
    const float* x     = (const float*)d_in[0];
    const int*   ei    = (const int*)d_in[1];
    const int*   batch = (const int*)d_in[2];
    const float* Wl[4] = {(const float*)d_in[3], (const float*)d_in[6],
                          (const float*)d_in[9], (const float*)d_in[12]};
    const float* bl[4] = {(const float*)d_in[4], (const float*)d_in[7],
                          (const float*)d_in[10], (const float*)d_in[13]};
    const float* Wr[4] = {(const float*)d_in[5], (const float*)d_in[8],
                          (const float*)d_in[11], (const float*)d_in[14]};
    const float* Wfc = (const float*)d_in[15];
    const float* bfc = (const float*)d_in[16];

    const int N = in_sizes[0] / 3;
    const int E = in_sizes[1] / 2;
    const int* src = ei;
    const int* dst = ei + E;

    // workspace carve-out (256B aligned)
    unsigned char* ws = (unsigned char*)d_ws;
    size_t off = 0;
    auto take = [&](size_t bytes) -> void* {
        void* p = ws + off;
        off += (bytes + 255) & ~(size_t)255;
        return p;
    };
    float* invc = (float*)take((size_t)N * 4);
    float* agg  = (float*)take((size_t)N * HID * 4);
    float* hA   = (float*)take((size_t)N * HID * 4);
    float* hB   = (float*)take((size_t)N * HID * 4);
    float* psum = (float*)take((size_t)NGR * HID * 4);
    float* pcnt = (float*)take((size_t)NGR * 4);
    __bf16* Wlp[4];
    __bf16* Wrp[4];
    for (int i = 0; i < 4; ++i) {
        Wlp[i] = (__bf16*)take(4096 * 2);
        Wrp[i] = (__bf16*)take(4096 * 2);
    }
    __bf16* Wfcp = (__bf16*)take(8192 * 2);

    // pack all weights into WMMA B-fragment layout
    for (int i = 0; i < 4; ++i) {
        int K = (i == 0) ? 3 : HID;
        pack_weight_kern<<<16, 256, 0, stream>>>(Wl[i], Wlp[i], K, HID);
        pack_weight_kern<<<16, 256, 0, stream>>>(Wr[i], Wrp[i], K, HID);
    }
    pack_weight_kern<<<32, 256, 0, stream>>>(Wfc, Wfcp, HID, OUTC);

    // degrees
    hipMemsetAsync(invc, 0, (size_t)N * 4, stream);
    int eb = (E + 255) / 256;
    deg_kern<<<eb, 256, 0, stream>>>(dst, invc, E);
    invdeg_kern<<<(N + 255) / 256, 256, 0, stream>>>(invc, N);

    int tiles = (N + 15) / 16;
    int gblocks = (tiles * 32 + 255) / 256;
    long long stot = (long long)E * 16;
    int sblocks = (int)((stot + 255) / 256);

    // layer 1 (Kin = 3)
    hipMemsetAsync(agg, 0, (size_t)N * HID * 4, stream);
    scatter3_kern<<<eb, 256, 0, stream>>>(x, src, dst, agg, E);
    sage_layer_kern<3><<<gblocks, 256, 0, stream>>>(x, agg, invc, Wlp[0],
                                                    Wrp[0], bl[0], hA, N);

    // layers 2..4 (Kin = 64), ping-pong hA/hB
    const float* cur = hA;
    float* nxt = hB;
    for (int i = 1; i < 4; ++i) {
        hipMemsetAsync(agg, 0, (size_t)N * HID * 4, stream);
        scatter64_kern<<<sblocks, 256, 0, stream>>>(cur, src, dst, agg, E);
        sage_layer_kern<HID><<<gblocks, 256, 0, stream>>>(
            cur, agg, invc, Wlp[i], Wrp[i], bl[i], nxt, N);
        float* t = (float*)cur;
        cur = nxt;
        nxt = t;
    }

    // global mean pool
    hipMemsetAsync(psum, 0, (size_t)NGR * HID * 4, stream);
    hipMemsetAsync(pcnt, 0, (size_t)NGR * 4, stream);
    long long ptot = (long long)N * 16;
    pool_kern<<<(int)((ptot + 255) / 256), 256, 0, stream>>>(cur, batch, psum,
                                                             pcnt, N);

    // FC + L2 normalize
    final_fc_kern<<<1, 256, 0, stream>>>(psum, pcnt, Wfcp, bfc, (float*)d_out);
}
